// VeterinaryMambaLM_79439715107078
// MI455X (gfx1250) — matrix operand order repaired and checked by reading, not compiled, not run on hardware
//
#include <hip/hip_runtime.h>
#include <stddef.h>

// ---------------- types for CDNA5 WMMA ----------------
typedef __attribute__((ext_vector_type(16))) __bf16 v16bf;
typedef __attribute__((ext_vector_type(8)))  __bf16 v8bf;
typedef __attribute__((ext_vector_type(8)))  float  v8f;

union AFrag { v16bf v; v8bf h[2]; };

// model dims
#define BATCH     8
#define SEQ       512
#define NTOK      (BATCH*SEQ)      // 4096
#define DMODEL    512
#define DSTATE    64
#define DCONV     4
#define DINNER    1024
#define NHEADS    8
#define HEADDIM   128
#define VOCAB     50257
#define DCONVCH   (DINNER + 2*DSTATE)        // 1152
#define DINPROJ   (2*DINNER + 2*DSTATE + NHEADS) // 2184

__device__ inline __bf16 f2bf(float f) {
    union { float f; unsigned u; } x; x.f = f;
    unsigned r = x.u + 0x7FFFu + ((x.u >> 16) & 1u);
    union { unsigned short s; __bf16 b; } y; y.s = (unsigned short)(r >> 16);
    return y.b;
}

__device__ inline float block_sum_256(float v, float* red) {
    int tid = threadIdx.x;
    red[tid] = v; __syncthreads();
    #pragma unroll
    for (int off = 128; off > 0; off >>= 1) {
        if (tid < off) red[tid] += red[tid + off];
        __syncthreads();
    }
    float r = red[0];
    __syncthreads();
    return r;
}

// ---------------- f32 -> bf16 convert ----------------
__global__ __launch_bounds__(256) void k_f32_to_bf16(const float* __restrict__ in,
                                                     __bf16* __restrict__ out, size_t n) {
    size_t stride = (size_t)gridDim.x * blockDim.x;
    for (size_t i = (size_t)blockIdx.x * blockDim.x + threadIdx.x; i < n; i += stride)
        out[i] = f2bf(in[i]);
}

// ---------------- embedding + pos + layernorm -> bf16 ----------------
__global__ __launch_bounds__(256) void k_embed_ln(const int* __restrict__ ids,
                                                  const float* __restrict__ emb,
                                                  const float* __restrict__ pos,
                                                  const float* __restrict__ g,
                                                  const float* __restrict__ b,
                                                  __bf16* __restrict__ hout) {
    __shared__ float red[256];
    int tkn = blockIdx.x;            // 0..4095
    int l   = tkn & (SEQ - 1);
    int id  = ids[tkn];
    int tid = threadIdx.x;
    float v[2];
    #pragma unroll
    for (int j = 0; j < 2; ++j) {
        int d = tid + j * 256;
        v[j] = emb[(size_t)id * DMODEL + d] + pos[(size_t)l * DMODEL + d];
    }
    float mu  = block_sum_256(v[0] + v[1], red) * (1.0f / DMODEL);
    float d0 = v[0] - mu, d1 = v[1] - mu;
    float var = block_sum_256(d0 * d0 + d1 * d1, red) * (1.0f / DMODEL);
    float rs  = rsqrtf(var + 1e-5f);
    #pragma unroll
    for (int j = 0; j < 2; ++j) {
        int d = tid + j * 256;
        float dv = (j == 0 ? d0 : d1);
        hout[(size_t)tkn * DMODEL + d] = f2bf(dv * rs * g[d] + b[d]);
    }
}

// ---------------- generic bf16 WMMA GEMM: C[M,N] = A[M,K] * W[N,K]^T (+bias) ----------------
// one wave computes a 16x64 strip (4 N-tiles, A-fragment reused)
__global__ __launch_bounds__(256) void k_gemm_bf16_wmma(const __bf16* __restrict__ A,
                                                        const __bf16* __restrict__ W,
                                                        const float* __restrict__ bias,
                                                        float* __restrict__ C,
                                                        int M, int N, int K,
                                                        int tiles_m, int tiles_n4) {
    int wave = blockIdx.x * (blockDim.x >> 5) + (threadIdx.x >> 5);
    if (wave >= tiles_m * tiles_n4) return;   // wave-uniform: EXEC stays all-1s for WMMA
    int tm  = wave % tiles_m;
    int tn4 = wave / tiles_m;
    int lane  = threadIdx.x & 31;
    int mlan  = lane & 15;
    int khalf = lane >> 4;

    const __bf16* arow = A + (size_t)(tm * 16 + mlan) * K + khalf * 16;
    const __bf16* wrow[4];
    #pragma unroll
    for (int j = 0; j < 4; ++j) {
        int n = tn4 * 64 + j * 16 + mlan;
        int nc = n < N ? n : N - 1;           // clamp loads; stores guarded below
        wrow[j] = W + (size_t)nc * K + khalf * 16;
    }

    v8f acc[4];
    #pragma unroll
    for (int j = 0; j < 4; ++j) acc[j] = (v8f){0,0,0,0,0,0,0,0};

    for (int k0 = 0; k0 < K; k0 += 32) {
        AFrag a;
        a.h[0] = *(const v8bf*)(arow + k0);        // K = kb .. kb+7
        a.h[1] = *(const v8bf*)(arow + k0 + 8);    // K = kb+8 .. kb+15
        #pragma unroll
        for (int j = 0; j < 4; ++j) {
            v16bf bfrag = *(const v16bf*)(wrow[j] + k0);  // 16 contiguous K of row n
            acc[j] = __builtin_amdgcn_wmma_f32_16x16x32_bf16(
                         false, a.v, false, bfrag, (short)0, acc[j], false, false);
        }
    }

    int mbase = tm * 16 + khalf * 8;   // D VGPR r -> M = mbase + r
    #pragma unroll
    for (int j = 0; j < 4; ++j) {
        int n = tn4 * 64 + j * 16 + mlan;
        if (n < N) {
            float bv = bias ? bias[n] : 0.0f;
            #pragma unroll
            for (int r = 0; r < 8; ++r)
                C[(size_t)(mbase + r) * N + n] = acc[j][r] + bv;
        }
    }
}

// ---------------- depthwise conv (width 4) + SiLU ----------------
__global__ __launch_bounds__(256) void k_conv_silu(const float* __restrict__ zx,
                                                   const float* __restrict__ cw,
                                                   const float* __restrict__ cb,
                                                   float* __restrict__ xbc) {
    size_t idx = (size_t)blockIdx.x * blockDim.x + threadIdx.x;
    if (idx >= (size_t)NTOK * DCONVCH) return;
    int c = (int)(idx % DCONVCH);
    int t = (int)(idx / DCONVCH);
    int l = t & (SEQ - 1);
    float acc = cb[c];
    #pragma unroll
    for (int k = 0; k < DCONV; ++k) {
        int ls = l - (DCONV - 1) + k;
        if (ls >= 0)
            acc += zx[(size_t)(t - (DCONV - 1) + k) * DINPROJ + DINNER + c] * cw[c * DCONV + k];
    }
    xbc[idx] = acc / (1.0f + __expf(-acc));   // SiLU
}

// ---------------- dt = softplus(raw + dt_bias) ----------------
__global__ __launch_bounds__(256) void k_dt(const float* __restrict__ zx,
                                            const float* __restrict__ dtb,
                                            float* __restrict__ dt) {
    int idx = blockIdx.x * blockDim.x + threadIdx.x;
    if (idx >= NTOK * NHEADS) return;
    int h = idx & (NHEADS - 1);
    int t = idx >> 3;
    float v = zx[(size_t)t * DINPROJ + (DINPROJ - NHEADS) + h] + dtb[h];
    dt[idx] = (v > 20.0f) ? v : log1pf(__expf(v));
}

// ---------------- SSM scan: one block per (b,h), 256 thr, 32 states/thr ----------------
__global__ __launch_bounds__(256) void k_scan(const float* __restrict__ xbc,
                                              const float* __restrict__ dt,
                                              const float* __restrict__ A_log,
                                              const float* __restrict__ Dp,
                                              float* __restrict__ y) {
    __shared__ float sx[HEADDIM];
    __shared__ float sB[DSTATE];
    __shared__ float sC[DSTATE];
    __shared__ float sdt;
    int b = blockIdx.x >> 3;
    int h = blockIdx.x & 7;
    int tid = threadIdx.x;
    float A  = -__expf(A_log[h]);
    float Dh = Dp[h];
    int p = tid >> 1, half = tid & 1, n0 = half * 32;
    float st[32];
    #pragma unroll
    for (int i = 0; i < 32; ++i) st[i] = 0.0f;
    size_t tokbase = (size_t)b * SEQ;
    for (int t = 0; t < SEQ; ++t) {
        const float* row = xbc + (tokbase + t) * DCONVCH;
        __syncthreads();
        if (tid < 128)      sx[tid]        = row[h * HEADDIM + tid];
        else if (tid < 192) sB[tid - 128]  = row[DINNER + (tid - 128)];
        else                sC[tid - 192]  = row[DINNER + DSTATE + (tid - 192)];
        if (tid == 0)       sdt            = dt[(tokbase + t) * NHEADS + h];
        __syncthreads();
        float dtv  = sdt;
        float dA   = __expf(dtv * A);
        float xp   = sx[p];
        float coef = dtv * xp;
        float sum  = 0.0f;
        #pragma unroll
        for (int i = 0; i < 32; ++i) {
            st[i] = st[i] * dA + coef * sB[n0 + i];
            sum  += st[i] * sC[n0 + i];
        }
        sum += __shfl_xor(sum, 1, 32);      // combine the two n-halves (wave32)
        if (half == 0)
            y[(tokbase + t) * DINNER + h * HEADDIM + p] = sum + Dh * xp;
    }
}

// ---------------- y * silu(z), RMSNorm over D_INNER -> bf16 ----------------
__global__ __launch_bounds__(256) void k_gate_rms(const float* __restrict__ y,
                                                  const float* __restrict__ zx,
                                                  const float* __restrict__ w,
                                                  __bf16* __restrict__ out) {
    __shared__ float red[256];
    int t = blockIdx.x, tid = threadIdx.x;
    float u[4];
    float ss = 0.0f;
    #pragma unroll
    for (int j = 0; j < 4; ++j) {
        int e = tid + j * 256;
        float z = zx[(size_t)t * DINPROJ + e];
        float sil = z / (1.0f + __expf(-z));
        u[j] = y[(size_t)t * DINNER + e] * sil;
        ss += u[j] * u[j];
    }
    float tot = block_sum_256(ss, red);
    float scale = rsqrtf(tot * (1.0f / DINNER) + 1e-5f);
    #pragma unroll
    for (int j = 0; j < 4; ++j) {
        int e = tid + j * 256;
        out[(size_t)t * DINNER + e] = f2bf(u[j] * scale * w[e]);
    }
}

// ---------------- RMSNorm over D_MODEL -> bf16 ----------------
__global__ __launch_bounds__(256) void k_rms_out(const float* __restrict__ mo,
                                                 const float* __restrict__ w,
                                                 __bf16* __restrict__ out) {
    __shared__ float red[256];
    int t = blockIdx.x, tid = threadIdx.x;
    float u[2];
    float ss = 0.0f;
    #pragma unroll
    for (int j = 0; j < 2; ++j) {
        int d = tid + j * 256;
        u[j] = mo[(size_t)t * DMODEL + d];
        ss += u[j] * u[j];
    }
    float tot = block_sum_256(ss, red);
    float scale = rsqrtf(tot * (1.0f / DMODEL) + 1e-5f);
    #pragma unroll
    for (int j = 0; j < 2; ++j) {
        int d = tid + j * 256;
        out[(size_t)t * DMODEL + d] = f2bf(u[j] * scale * w[d]);
    }
}

// ---------------- launch ----------------
extern "C" void kernel_launch(void* const* d_in, const int* in_sizes, int n_in,
                              void* d_out, int out_size, void* d_ws, size_t ws_size,
                              hipStream_t stream) {
    const int*   ids     = (const int*)  d_in[0];
    const float* emb     = (const float*)d_in[1];
    const float* pos     = (const float*)d_in[2];
    const float* ln_g    = (const float*)d_in[3];
    const float* ln_b    = (const float*)d_in[4];
    const float* W_in    = (const float*)d_in[5];
    const float* conv_w  = (const float*)d_in[6];
    const float* conv_b  = (const float*)d_in[7];
    const float* dt_bias = (const float*)d_in[8];
    const float* A_log   = (const float*)d_in[9];
    const float* Dparam  = (const float*)d_in[10];
    const float* nin_w   = (const float*)d_in[11];
    const float* W_mout  = (const float*)d_in[12];
    const float* nout_w  = (const float*)d_in[13];
    const float* W_vocab = (const float*)d_in[14];
    const float* b_vocab = (const float*)d_in[15];
    float* logits = (float*)d_out;

    char* ws = (char*)d_ws;
    size_t off = 0;
    auto alloc = [&](size_t bytes) { size_t o = off; off = (off + bytes + 255) & ~(size_t)255; return o; };
    __bf16* winb = (__bf16*)(ws + alloc((size_t)DINPROJ * DMODEL * 2));
    __bf16* wmob = (__bf16*)(ws + alloc((size_t)DMODEL * DINNER * 2));
    __bf16* wvob = (__bf16*)(ws + alloc((size_t)VOCAB * DMODEL * 2));
    __bf16* hbf  = (__bf16*)(ws + alloc((size_t)NTOK * DMODEL * 2));
    float*  zx   = (float*) (ws + alloc((size_t)NTOK * DINPROJ * 4));
    float*  xbc  = (float*) (ws + alloc((size_t)NTOK * DCONVCH * 4));
    float*  dtw  = (float*) (ws + alloc((size_t)NTOK * NHEADS * 4));
    float*  yb   = (float*) (ws + alloc((size_t)NTOK * DINNER * 4));
    __bf16* ynb  = (__bf16*)(ws + alloc((size_t)NTOK * DINNER * 2));
    float*  mob  = (float*) (ws + alloc((size_t)NTOK * DMODEL * 4));
    __bf16* nob  = (__bf16*)(ws + alloc((size_t)NTOK * DMODEL * 2));

    // 1) weight conversions to bf16
    k_f32_to_bf16<<<4096, 256, 0, stream>>>(W_in,    winb, (size_t)DINPROJ * DMODEL);
    k_f32_to_bf16<<<4096, 256, 0, stream>>>(W_mout,  wmob, (size_t)DMODEL * DINNER);
    k_f32_to_bf16<<<8192, 256, 0, stream>>>(W_vocab, wvob, (size_t)VOCAB * DMODEL);

    // 2) embedding + pos + LN -> bf16
    k_embed_ln<<<NTOK, 256, 0, stream>>>(ids, emb, pos, ln_g, ln_b, hbf);

    // 3) in-projection GEMM: zx[4096,2184] = hbf @ W_in^T
    {
        int tiles_m = NTOK / 16, tiles_n4 = (DINPROJ + 63) / 64;
        int waves = tiles_m * tiles_n4;
        k_gemm_bf16_wmma<<<(waves + 7) / 8, 256, 0, stream>>>(
            hbf, winb, nullptr, zx, NTOK, DINPROJ, DMODEL, tiles_m, tiles_n4);
    }

    // 4) conv + silu ; dt softplus
    {
        size_t n = (size_t)NTOK * DCONVCH;
        k_conv_silu<<<(int)((n + 255) / 256), 256, 0, stream>>>(zx, conv_w, conv_b, xbc);
        k_dt<<<(NTOK * NHEADS + 255) / 256, 256, 0, stream>>>(zx, dt_bias, dtw);
    }

    // 5) SSM scan (64 blocks = B*NHEADS)
    k_scan<<<BATCH * NHEADS, 256, 0, stream>>>(xbc, dtw, A_log, Dparam, yb);

    // 6) gate + RMSNorm -> bf16
    k_gate_rms<<<NTOK, 256, 0, stream>>>(yb, zx, nin_w, ynb);

    // 7) out-projection GEMM: mo[4096,512] = ynb @ W_mout^T
    {
        int tiles_m = NTOK / 16, tiles_n4 = (DMODEL + 63) / 64;
        int waves = tiles_m * tiles_n4;
        k_gemm_bf16_wmma<<<(waves + 7) / 8, 256, 0, stream>>>(
            ynb, wmob, nullptr, mob, NTOK, DMODEL, DINNER, tiles_m, tiles_n4);
    }

    // 8) RMSNorm -> bf16
    k_rms_out<<<NTOK, 256, 0, stream>>>(mob, nout_w, nob);

    // 9) vocab GEMM: logits[4096,50257] = nob @ W_vocab^T + b_vocab
    {
        int tiles_m = NTOK / 16, tiles_n4 = (VOCAB + 63) / 64;
        int waves = tiles_m * tiles_n4;
        k_gemm_bf16_wmma<<<(waves + 7) / 8, 256, 0, stream>>>(
            nob, wvob, b_vocab, logits, NTOK, VOCAB, DMODEL, tiles_m, tiles_n4);
    }
}
